// LSTM_model_55138790146616
// MI455X (gfx1250) — matrix operand order repaired and checked by reading, not compiled, not run on hardware
//
#include <hip/hip_runtime.h>
#include <hip/hip_fp16.h>

// ---------------------------------------------------------------------------
// Fused LSTM for MI455X (gfx1250, wave32, WMMA).
//   grid  = B/16 = 32 workgroups, one per 16-row batch tile (WMMA M-tile)
//   block = 25 waves (800 threads); wave w owns gate-column tile N=[16w,16w+16)
// Per timestep (serial loop of 1024 inside the kernel):
//   z = bias + x_t @ Wx  (3x v_wmma_f32_16x16x4_f32, exact fp32, K=12)
//         + h   @ Wh     (4x v_wmma_f32_16x16x32_f16, K padded 100->128)
//   gates (v_exp_f32 / v_tanh_f32) -> c,h update -> h stored f16 in LDS
//   out_t = h @ Wd + bd  (fused, K=100, written straight to d_out)
// Wh/Wx live in VGPRs as pre-swizzled B-fragments for the entire kernel.
// ---------------------------------------------------------------------------

typedef float     v2f  __attribute__((ext_vector_type(2)));
typedef float     v8f  __attribute__((ext_vector_type(8)));
typedef _Float16  v16h __attribute__((ext_vector_type(16)));

#define B_   512
#define T_   1024
#define D_   12
#define H_   100
#define G4H  400     // 4*H
#define KP   128     // K padding for h (100 -> 128, zero-filled)
#define WAVES 25
#define THREADS (WAVES * 32)

#if __has_builtin(__builtin_amdgcn_wmma_f32_16x16x4_f32)
#define HAVE_WMMA_F32X4 1
#else
#define HAVE_WMMA_F32X4 0
#endif

union V16HU { v16h h; uint4 u[2]; };

__device__ __forceinline__ float sigmoid_fast(float x) {
  // v_exp_f32 + v_rcp_f32, no IEEE-div fixup on the critical path
  return __builtin_amdgcn_rcpf(1.f + __expf(-x));
}

__device__ __forceinline__ float tanh_fast(float x) {
#if __has_builtin(__builtin_amdgcn_tanhf)
  return __builtin_amdgcn_tanhf(x);       // native V_TANH_F32 (CDNA5 TRANS op)
#else
  // branchless: saturates correctly as exp(2x) -> 0 / inf
  const float e2 = __expf(2.f * x);
  return 1.f - 2.f * __builtin_amdgcn_rcpf(e2 + 1.f);
#endif
}

__global__ __launch_bounds__(THREADS) void lstm_fused(
    const float* __restrict__ x,  const float* __restrict__ Wx,
    const float* __restrict__ Wh, const float* __restrict__ b,
    const float* __restrict__ Wd, const float* __restrict__ bd,
    float* __restrict__ out)
{
  __shared__ _Float16 hbuf[16][KP];    // h state, f16 row-major (A-frag source)
  __shared__ float    zbuf[G4H][16];   // gate pre-activations, TRANSPOSED:
                                       //   [col][row] so C-frag rows are contiguous
  __shared__ float    sWd[H_][D_];
  __shared__ float    sbd[D_];
#if !HAVE_WMMA_F32X4
  __shared__ float    sWx[D_][G4H];
  __shared__ float    sb[G4H];
#endif

  const int tid  = threadIdx.x;
  const int wave = tid >> 5;
  const int lane = tid & 31;
  const int l16  = lane & 15;
  const int lhi  = lane >> 4;              // half-wave select (0/1)
  const int rowBase = blockIdx.x * 16;

  // ---- one-time setup: cache small tensors, zero h state -------------------
  for (int i = tid; i < H_ * D_; i += THREADS) sWd[i / D_][i % D_] = Wd[i];
  if (tid < D_) sbd[tid] = bd[tid];
  for (int i = tid; i < 16 * KP; i += THREADS) (&hbuf[0][0])[i] = (_Float16)0.f;
#if !HAVE_WMMA_F32X4
  for (int i = tid; i < D_ * G4H; i += THREADS) (&sWx[0][0])[i] = Wx[i];
  for (int i = tid; i < G4H; i += THREADS) sb[i] = b[i];
#endif

  // ---- preload this wave's Wh B-fragments (f16 32x16 per K-tile) -----------
  // ISA 7.12.2: VGPR j, lanes 0-15 hold K=2j,2j+1 (+8 for j>=4) col=lane; lanes 16-31 K+=8.
  const int nt  = wave;                    // N-tile owned by this wave
  const int col = nt * 16 + l16;           // gate column of this lane
  const float biasv = b[col];
  v16h whf[4];
  for (int kt = 0; kt < 4; ++kt) {
    const int kb = kt * 32 + lhi * 8;
    v16h f;
    for (int j = 0; j < 16; ++j) {
      const int k = kb + (j < 8 ? j : j + 8);
      f[j] = (k < H_) ? (_Float16)Wh[k * G4H + col] : (_Float16)0.f;
    }
    whf[kt] = f;
  }
#if HAVE_WMMA_F32X4
  // Wx B-fragments (f32 4x16): VGPR0 K=kt*4+lhi*2, VGPR1 = +1
  v2f wxf[3];
  for (int kt = 0; kt < 3; ++kt) {
    const int k0 = kt * 4 + lhi * 2;
    v2f g;
    g[0] = Wx[(k0    ) * G4H + col];
    g[1] = Wx[(k0 + 1) * G4H + col];
    wxf[kt] = g;
  }
#endif
  __syncthreads();

  // per-thread cell state: row m = tid%16, two consecutive hidden cols
  const int em    = tid & 15;
  const int cbase = (tid >> 4) * 2;        // tid>>4 in [0,50) -> cbase < 100
  float cst[2] = {0.f, 0.f};

  const float* xrow = x + (size_t)(rowBase + l16) * T_ * D_;

  for (int t = 0; t < T_; ++t) {
    if (wave == 0 && t + 2 < T_)           // one 128B line covers ~2.6 steps
      __builtin_prefetch(xrow + (size_t)(t + 2) * D_, 0, 3);  // WGP-scope prefetch

    // A-fragments of h from LDS: per K-tile two 16B chunks (ds_load_b128)
    // lanes 0-15: K = kt*32 + {0..7, 16..23}; lanes 16-31: +8
    V16HU ha[4];
#pragma unroll
    for (int kt = 0; kt < 4; ++kt) {
      const int kb = kt * 32 + lhi * 8;
      ha[kt].u[0] = *(const uint4*)&hbuf[l16][kb];
      ha[kt].u[1] = *(const uint4*)&hbuf[l16][kb + 16];
    }

    // ---- z tile: bias + x@Wx + h@Wh via WMMA -------------------------------
    v8f acc;
#if HAVE_WMMA_F32X4
#pragma unroll
    for (int r = 0; r < 8; ++r) acc[r] = biasv;
    {
      // A-fragments of x (f32 16x4): lane holds K = kt*4 + lhi*2, +1
      v2f xa[3];
#pragma unroll
      for (int kt = 0; kt < 3; ++kt) {
        const float2 v = *(const float2*)&xrow[(size_t)t * D_ + kt * 4 + lhi * 2];
        v2f g; g[0] = v.x; g[1] = v.y;
        xa[kt] = g;
      }
#pragma unroll
      for (int kt = 0; kt < 3; ++kt)
        acc = __builtin_amdgcn_wmma_f32_16x16x4_f32(
            false, xa[kt], false, wxf[kt], (short)0, acc, false, false);
    }
#else
#pragma unroll
    for (int r = 0; r < 8; ++r) acc[r] = 0.f;
#endif
#pragma unroll
    for (int kt = 0; kt < 4; ++kt)
      acc = __builtin_amdgcn_wmma_f32_16x16x32_f16(
          false, ha[kt].h, false, whf[kt], (short)0, acc, false, false);

    // C/D layout: VGPR r -> M = lhi*8 + r, N = col.  With zbuf transposed the
    // 8 values per lane are contiguous -> two ds_store_b128.
    {
      float4 lo = make_float4(acc[0], acc[1], acc[2], acc[3]);
      float4 hi = make_float4(acc[4], acc[5], acc[6], acc[7]);
      *(float4*)&zbuf[col][lhi * 8 + 0] = lo;
      *(float4*)&zbuf[col][lhi * 8 + 4] = hi;
    }
    __syncthreads();

    // ---- gates + cell/hidden update (row em, cols cbase..cbase+1) ----------
#pragma unroll
    for (int j = 0; j < 2; ++j) {
      const int c = cbase + j;
      float zi = zbuf[c        ][em];
      float zf = zbuf[H_ + c   ][em];
      float zg = zbuf[2 * H_ + c][em];
      float zo = zbuf[3 * H_ + c][em];
#if !HAVE_WMMA_F32X4
      // fallback: add bias + x_t@Wx here via VALU FMA
      const float* xr = x + ((size_t)(rowBase + em) * T_ + t) * D_;
      float xg0 = sb[c], xg1 = sb[H_ + c], xg2 = sb[2 * H_ + c], xg3 = sb[3 * H_ + c];
      for (int k = 0; k < D_; ++k) {
        const float xv = xr[k];
        xg0 += xv * sWx[k][c];
        xg1 += xv * sWx[k][H_ + c];
        xg2 += xv * sWx[k][2 * H_ + c];
        xg3 += xv * sWx[k][3 * H_ + c];
      }
      zi += xg0; zf += xg1; zg += xg2; zo += xg3;
#endif
      const float ig = sigmoid_fast(zi);
      const float fg = sigmoid_fast(zf);
      const float gg = tanh_fast(zg);
      const float og = sigmoid_fast(zo);
      const float cn = fmaf(fg, cst[j], ig * gg);
      cst[j] = cn;
      hbuf[em][c] = (_Float16)(og * tanh_fast(cn));
    }
    __syncthreads();

    // ---- fused output projection: out[b,t,:] = h @ Wd + bd -----------------
    for (int idx = tid; idx < 16 * D_; idx += THREADS) {
      const int m = idx / D_, d = idx % D_;
      float s = sbd[d];
#pragma unroll 4
      for (int c = 0; c < H_; ++c) s += (float)hbuf[m][c] * sWd[c][d];
      out[((size_t)(rowBase + m) * T_ + t) * D_ + d] = s;
    }
    // next iteration's first __syncthreads separates these hbuf reads from the
    // next step's hbuf writes; zbuf reuse is separated by the barrier above.
  }
}

extern "C" void kernel_launch(void* const* d_in, const int* in_sizes, int n_in,
                              void* d_out, int out_size, void* d_ws, size_t ws_size,
                              hipStream_t stream) {
  (void)in_sizes; (void)n_in; (void)out_size; (void)d_ws; (void)ws_size;
  const float* x  = (const float*)d_in[0];
  const float* Wx = (const float*)d_in[1];
  const float* Wh = (const float*)d_in[2];
  const float* b  = (const float*)d_in[3];
  const float* Wd = (const float*)d_in[4];
  const float* bd = (const float*)d_in[5];
  float* out = (float*)d_out;
  lstm_fused<<<dim3(B_ / 16), dim3(THREADS), 0, stream>>>(x, Wx, Wh, b, Wd, bd, out);
}
